// TransformerDecoderLayer_19851338842359
// MI455X (gfx1250) — matrix operand order repaired and checked by reading, not compiled, not run on hardware
//
#include <hip/hip_runtime.h>
#include <hip/hip_bf16.h>

// ---------------------------------------------------------------------------
// MI455X (gfx1250) TransformerDecoderLayer
//   - all GEMMs via v_wmma_f32_16x16x32_bf16 (fp32 inputs converted inline)
//   - wave32: each wave owns one 16x16 output tile, 8 waves per 256-thr block
//   - GEMM templated on TRANSB/RELU so the K-loop is branch-free
// ---------------------------------------------------------------------------

typedef __bf16 bf16;
typedef __attribute__((ext_vector_type(16))) __bf16 v16bf;
typedef __attribute__((ext_vector_type(8)))  float  v8f;

#define LQ   256
#define BZ   32
#define LV   4096
#define DM   256
#define DFF  2048
#define NH   8
#define NK   4
#define HD   32
#define NQROWS (LQ * BZ)   /* 8192   */
#define NVROWS (LV * BZ)   /* 131072 */

// ---------------------------------------------------------------------------
// Generic batched GEMM:  C = A * op(B) + bias   (fp32 in/out, bf16 WMMA core)
//   A: [M x Kd], row stride lda
//   TRANSB == 0: B stored [N x Kd] (weights, compute A * B^T)
//   TRANSB == 1: B stored [Kd x N] (compute A * B)
//   batch z decomposes as z1 = z / batch2, z2 = z % batch2 with independent
//   element-offset strides for A/B/C (covers per-(b,h) attention batching).
// ---------------------------------------------------------------------------
template <int TRANSB, int RELU>
__global__ __launch_bounds__(256) void gemm_bf16_wmma(
    const float* __restrict__ A, long lda, long sA1, long sA2,
    const float* __restrict__ B, long ldb, long sB1, long sB2,
    const float* __restrict__ bias,
    float* __restrict__ C, long ldc, long sC1, long sC2,
    int M, int N, int Kd, int batch2)
{
  const int z  = blockIdx.z;
  const int z1 = z / batch2;
  const int z2 = z - z1 * batch2;
  A += z1 * sA1 + z2 * sA2;
  B += z1 * sB1 + z2 * sB2;
  C += z1 * sC1 + z2 * sC2;

  const int wave = threadIdx.x >> 5;
  const int lane = threadIdx.x & 31;
  const int row0 = blockIdx.y * 128 + wave * 16;
  const int col0 = blockIdx.x * 16;
  if (row0 >= M) return;

  const int mrow  = lane & 15;   // A-fragment row within tile
  const int hi    = lane >> 4;   // lane half select
  const int koffA = hi * 16;     // lanes 16-31 hold K+16 block for A
  const int koffB = hi * 16;     // lanes 16-31 hold K+16 block for B (same order as A)
  const int ncol  = lane & 15;   // B/C column within tile

  // Per-lane base pointers, advanced additively each K-step (no per-iter mul).
  const float* aptr = A + (long)(row0 + mrow) * lda + koffA;
  const float* bptr = TRANSB
      ? (B + (long)koffB * ldb + col0 + ncol)
      : (B + (long)(col0 + ncol) * ldb + koffB);
  const long bstep = TRANSB ? (long)32 * ldb : 32;

  v8f acc = {};
  for (int k0 = 0; k0 < Kd; k0 += 32) {
    v16bf afrag, bfrag;
#pragma unroll
    for (int i = 0; i < 8; ++i) {
      // A 16x32 bf16 layout: vgpr i -> K = 2i,2i+1 (lanes 0-15), +16 (16-31)
      const int kk = 2 * i;
      const float2 av = *(const float2*)(aptr + kk);
      afrag[2 * i]     = (bf16)av.x;
      afrag[2 * i + 1] = (bf16)av.y;
    }
#pragma unroll
    for (int i = 0; i < 8; ++i) {
      // B 32x16 bf16 layout: vgpr i -> K = 2i,2i+1 (lanes 0-15), +16 (16-31)
      float b0, b1;
      if (TRANSB) {
        b0 = bptr[(long)(2 * i) * ldb];
        b1 = bptr[(long)(2 * i + 1) * ldb];
      } else {
        const float2 bv = *(const float2*)(bptr + 2 * i);
        b0 = bv.x; b1 = bv.y;
      }
      bfrag[2 * i]     = (bf16)b0;
      bfrag[2 * i + 1] = (bf16)b1;
    }
    acc = __builtin_amdgcn_wmma_f32_16x16x32_bf16(
        /*neg_a=*/false, afrag, /*neg_b=*/false, bfrag,
        /*c_mod=*/(short)0, acc, /*reuse_a=*/false, /*reuse_b=*/false);
    aptr += 32;
    bptr += bstep;
  }

  const float bval = bias ? bias[col0 + ncol] : 0.f;
#pragma unroll
  for (int r = 0; r < 8; ++r) {
    const int mo = hi * 8 + r;          // C layout: lanes 16-31 hold M+8
    float v = acc[r] + bval;
    if (RELU) v = fmaxf(v, 0.f);
    C[(long)(row0 + mo) * ldc + col0 + ncol] = v;
  }
}

// ---------------------------------------------------------------------------
// q = tgt + query_pos ; qn = q / max(||q||, eps)   (one 256-thr block per row)
// ---------------------------------------------------------------------------
__global__ __launch_bounds__(256) void qprep_k(const float* __restrict__ tgt,
                                               const float* __restrict__ qp,
                                               float* __restrict__ q,
                                               float* __restrict__ qn)
{
  __shared__ float red[256];
  const int r = blockIdx.x, t = threadIdx.x;
  const long base = (long)r * DM;
  const float v = tgt[base + t] + qp[base + t];
  q[base + t] = v;
  red[t] = v * v; __syncthreads();
  for (int s = 128; s > 0; s >>= 1) { if (t < s) red[t] += red[t + s]; __syncthreads(); }
  const float nrm = sqrtf(red[0]);
  qn[base + t] = v / fmaxf(nrm, 1e-8f);
}

// ---------------------------------------------------------------------------
// graph mask + scaled softmax over attention scores (row per block)
// blockIdx.x = (b*NH + h)*LQ + qi ; thread = key index
// ---------------------------------------------------------------------------
__global__ __launch_bounds__(256) void mask_softmax_k(float* __restrict__ attn,
                                                      const float* __restrict__ cosb,
                                                      const float* __restrict__ ref)
{
  __shared__ float red[256];
  const int idx = blockIdx.x;
  const int qi  = idx & (LQ - 1);
  const int bh  = idx >> 8;
  const int b   = bh >> 3;
  const int k   = threadIdx.x;

  const float cq = ref[((long)b * LQ + qi) * 2 + 0];
  const float wq = ref[((long)b * LQ + qi) * 2 + 1];
  const float ck = ref[((long)b * LQ + k) * 2 + 0];
  const float wk = ref[((long)b * LQ + k) * 2 + 1];
  const float sq = cq - wq * 0.5f, eq = cq + wq * 0.5f;
  const float sk = ck - wk * 0.5f, ek = ck + wk * 0.5f;
  const float inter = fmaxf(fminf(eq, ek) - fmaxf(sq, sk), 0.f);
  const float uni   = (eq - sq) + (ek - sk) - inter;
  const float iou   = inter / fmaxf(uni, 1e-8f);
  const float cv    = cosb[((long)b * LQ + qi) * LQ + k];

  bool adj = (qi == k);
  if (iou <= 0.2f) adj = true;
  if (cv  <= 0.2f) adj = false;       // mask = !adj

  const long p = (long)idx * LQ + k;
  float sc = attn[p] * 0.17677669529663687f;  // 1/sqrt(hd=32)
  if (!adj) sc = -3.0e38f;

  red[k] = sc; __syncthreads();
  for (int s = 128; s > 0; s >>= 1) { if (k < s) red[k] = fmaxf(red[k], red[k + s]); __syncthreads(); }
  const float mx = red[0]; __syncthreads();
  const float ex = adj ? __expf(sc - mx) : 0.f;
  red[k] = ex; __syncthreads();
  for (int s = 128; s > 0; s >>= 1) { if (k < s) red[k] += red[k + s]; __syncthreads(); }
  attn[p] = ex / fmaxf(red[0], 1e-20f);
}

// ---------------------------------------------------------------------------
// out = LN(a + b) * g + be   (one 256-thr block per row, d=256)
// ---------------------------------------------------------------------------
__global__ __launch_bounds__(256) void ln_add_k(const float* __restrict__ a,
                                                const float* __restrict__ b,
                                                const float* __restrict__ g,
                                                const float* __restrict__ be,
                                                float* __restrict__ out)
{
  __shared__ float red[256];
  const int r = blockIdx.x, t = threadIdx.x;
  const long base = (long)r * DM;
  const float v = a[base + t] + b[base + t];
  red[t] = v; __syncthreads();
  for (int s = 128; s > 0; s >>= 1) { if (t < s) red[t] += red[t + s]; __syncthreads(); }
  const float mean = red[0] * (1.f / DM); __syncthreads();
  const float dv = v - mean;
  red[t] = dv * dv; __syncthreads();
  for (int s = 128; s > 0; s >>= 1) { if (t < s) red[t] += red[t + s]; __syncthreads(); }
  const float var = red[0] * (1.f / DM);
  out[base + t] = dv * rsqrtf(var + 1e-5f) * g[t] + be[t];
}

__global__ void add_inplace_k(float* __restrict__ a, const float* __restrict__ b, long n)
{
  long i = (long)blockIdx.x * blockDim.x + threadIdx.x;
  const long stride = (long)gridDim.x * blockDim.x;
  for (; i < n; i += stride) a[i] += b[i];
}

__global__ void vzero_pad_k(float* __restrict__ v, const unsigned char* __restrict__ mask, long n)
{
  long i = (long)blockIdx.x * blockDim.x + threadIdx.x;
  const long stride = (long)gridDim.x * blockDim.x;
  for (; i < n; i += stride) {
    const long row = i >> 8;               // / DM
    const long lv  = row / BZ;
    const long b   = row - lv * BZ;
    if (mask[b * LV + lv]) v[i] = 0.f;
  }
}

// ---------------------------------------------------------------------------
// deformable sampling: one wave per (b,h,l); thread = channel c in [0,32)
// aw softmax over K=4, 1-D linear grid sample (align_corners, zero-pad)
// ---------------------------------------------------------------------------
__global__ __launch_bounds__(32) void deform_sample_k(const float* __restrict__ vproj,
                                                      const float* __restrict__ offr,
                                                      const float* __restrict__ awr,
                                                      const float* __restrict__ ref,
                                                      float* __restrict__ out)
{
  const int idx = blockIdx.x;          // (b*NH + h)*LQ + l
  const int l  = idx & (LQ - 1);
  const int bh = idx >> 8;
  const int h  = bh & (NH - 1);
  const int b  = bh >> 3;
  const int c  = threadIdx.x;

  const long qrow = (long)l * BZ + b;
  const float rc = ref[((long)b * LQ + l) * 2 + 0];
  const float rw = ref[((long)b * LQ + l) * 2 + 1];

  float awv[NK];
  float mx = -3.4e38f;
#pragma unroll
  for (int k = 0; k < NK; ++k) {
    awv[k] = awr[qrow * (NH * NK) + h * NK + k];
    mx = fmaxf(mx, awv[k]);
  }
  float asum = 0.f;
#pragma unroll
  for (int k = 0; k < NK; ++k) { awv[k] = __expf(awv[k] - mx); asum += awv[k]; }
  const float ainv = 1.f / asum;

  float accv = 0.f;
#pragma unroll
  for (int k = 0; k < NK; ++k) {
    const float o   = offr[qrow * (NH * NK) + h * NK + k];
    const float pos = rc + o * (1.f / NK) * rw * 0.5f;
    const float gx  = pos * 2.f - 1.f;
    const float x   = (gx + 1.f) * (float)(LV - 1) * 0.5f;
    const float x0  = floorf(x);
    const float w1  = x - x0;
    const int   i0  = (int)x0;
    float v0 = 0.f, v1 = 0.f;
    if (i0 >= 0 && i0 <= LV - 1)
      v0 = vproj[((long)i0 * BZ + b) * DM + h * HD + c];
    if (i0 + 1 >= 0 && i0 + 1 <= LV - 1)
      v1 = vproj[((long)(i0 + 1) * BZ + b) * DM + h * HD + c];
    accv += (v0 * (1.f - w1) + v1 * w1) * (awv[k] * ainv);
  }
  out[qrow * DM + h * HD + c] = accv;
}

// ---------------------------------------------------------------------------
// workspace layout (float elements). "BIG" region is time-shared:
//   phase1: attn scores/probs (16.7M)  phase2: vproj (33.5M)  phase3: FFN hid
// ---------------------------------------------------------------------------
static const size_t OFF_Q    = 0;
static const size_t OFF_QN   = OFF_Q    + (size_t)NQROWS * DM;
static const size_t OFF_QH   = OFF_QN   + (size_t)NQROWS * DM;
static const size_t OFF_KH   = OFF_QH   + (size_t)NQROWS * DM;
static const size_t OFF_VH   = OFF_KH   + (size_t)NQROWS * DM;
static const size_t OFF_COS  = OFF_VH   + (size_t)NQROWS * DM;
static const size_t OFF_REL  = OFF_COS  + (size_t)BZ * LQ * LQ;
static const size_t OFF_X1   = OFF_REL  + (size_t)NQROWS * DM;
static const size_t OFF_OFFR = OFF_X1   + (size_t)NQROWS * DM;
static const size_t OFF_AWR  = OFF_OFFR + (size_t)NQROWS * (NH * NK);
static const size_t OFF_SAMP = OFF_AWR  + (size_t)NQROWS * (NH * NK);
static const size_t OFF_T2C  = OFF_SAMP + (size_t)NQROWS * DM;
static const size_t OFF_X2   = OFF_T2C  + (size_t)NQROWS * DM;
static const size_t OFF_FFN  = OFF_X2   + (size_t)NQROWS * DM;
static const size_t OFF_BIG  = OFF_FFN  + (size_t)NQROWS * DM;

static inline void launch_gemm(hipStream_t s,
                               const float* A, long lda, long sA1, long sA2,
                               const float* B, long ldb, long sB1, long sB2,
                               const float* bias,
                               float* C, long ldc, long sC1, long sC2,
                               int M, int N, int Kd,
                               int batches, int batch2, int transB, int relu)
{
  dim3 grid((unsigned)(N / 16), (unsigned)((M + 127) / 128), (unsigned)batches);
  if (transB)
    gemm_bf16_wmma<1, 0><<<grid, 256, 0, s>>>(A, lda, sA1, sA2, B, ldb, sB1, sB2,
                                              bias, C, ldc, sC1, sC2, M, N, Kd, batch2);
  else if (relu)
    gemm_bf16_wmma<0, 1><<<grid, 256, 0, s>>>(A, lda, sA1, sA2, B, ldb, sB1, sB2,
                                              bias, C, ldc, sC1, sC2, M, N, Kd, batch2);
  else
    gemm_bf16_wmma<0, 0><<<grid, 256, 0, s>>>(A, lda, sA1, sA2, B, ldb, sB1, sB2,
                                              bias, C, ldc, sC1, sC2, M, N, Kd, batch2);
}

extern "C" void kernel_launch(void* const* d_in, const int* in_sizes, int n_in,
                              void* d_out, int out_size, void* d_ws, size_t ws_size,
                              hipStream_t stream)
{
  (void)in_sizes; (void)n_in; (void)out_size; (void)ws_size;

  const float* tgt     = (const float*)d_in[0];
  const float* memory  = (const float*)d_in[1];
  const unsigned char* padmask = (const unsigned char*)d_in[2];
  const float* qpos    = (const float*)d_in[3];
  const float* ref     = (const float*)d_in[4];
  const float* Wq  = (const float*)d_in[5];
  const float* Wk  = (const float*)d_in[6];
  const float* Wv  = (const float*)d_in[7];
  const float* Woff= (const float*)d_in[8];
  const float* Waw = (const float*)d_in[9];
  const float* Wvp = (const float*)d_in[10];
  const float* Wop = (const float*)d_in[11];
  const float* W1  = (const float*)d_in[12];
  const float* W2  = (const float*)d_in[13];
  const float* bq  = (const float*)d_in[14];
  const float* bk  = (const float*)d_in[15];
  const float* bv  = (const float*)d_in[16];
  const float* baw = (const float*)d_in[17];
  const float* bvp = (const float*)d_in[18];
  const float* bop = (const float*)d_in[19];
  const float* bf1 = (const float*)d_in[20];
  const float* bf2 = (const float*)d_in[21];
  const float* boff= (const float*)d_in[22];
  const float* g1  = (const float*)d_in[23];
  const float* g2  = (const float*)d_in[24];
  const float* g3  = (const float*)d_in[25];
  const float* be1 = (const float*)d_in[26];
  const float* be2 = (const float*)d_in[27];
  const float* be3 = (const float*)d_in[28];

  float* ws    = (float*)d_ws;
  float* q     = ws + OFF_Q;
  float* qn    = ws + OFF_QN;
  float* Qh    = ws + OFF_QH;
  float* Kh    = ws + OFF_KH;
  float* Vh    = ws + OFF_VH;
  float* cosb  = ws + OFF_COS;
  float* rel   = ws + OFF_REL;
  float* x1    = ws + OFF_X1;
  float* offr  = ws + OFF_OFFR;
  float* awr   = ws + OFF_AWR;
  float* samp  = ws + OFF_SAMP;
  float* t2c   = ws + OFF_T2C;
  float* x2    = ws + OFF_X2;
  float* ffn   = ws + OFF_FFN;
  float* attn  = ws + OFF_BIG;   // phase 1
  float* vproj = ws + OFF_BIG;   // phase 2 (after attn dead)
  float* hid   = ws + OFF_BIG;   // phase 3 (after vproj dead)
  float* out   = (float*)d_out;

  const long LLQQ = (long)LQ * LQ;             // 65536
  const long SBH  = (long)NH * LLQQ;           // 524288

  // ---- relation attention ------------------------------------------------
  qprep_k<<<NQROWS, 256, 0, stream>>>(tgt, qpos, q, qn);

  // QKV projections: [8192x256] x [256x256]^T
  launch_gemm(stream, q, DM, 0, 0, Wq, DM, 0, 0, bq, Qh, DM, 0, 0,
              NQROWS, DM, DM, 1, 1, 0, 0);
  launch_gemm(stream, q, DM, 0, 0, Wk, DM, 0, 0, bk, Kh, DM, 0, 0,
              NQROWS, DM, DM, 1, 1, 0, 0);
  launch_gemm(stream, q, DM, 0, 0, Wv, DM, 0, 0, bv, Vh, DM, 0, 0,
              NQROWS, DM, DM, 1, 1, 0, 0);

  // cosine graph: per-batch qn * qn^T  (rows strided by bz*d)
  launch_gemm(stream, qn, BZ * DM, DM, 0, qn, BZ * DM, DM, 0, nullptr,
              cosb, LQ, LLQQ, 0, LQ, LQ, DM, BZ, 1, 0, 0);

  // scores = Qh * Kh^T per (b,h): Kd = hd = 32 (single WMMA step)
  launch_gemm(stream, Qh, BZ * DM, DM, HD, Kh, BZ * DM, DM, HD, nullptr,
              attn, LQ, SBH, LLQQ, LQ, LQ, HD, BZ * NH, NH, 0, 0);

  mask_softmax_k<<<BZ * NH * LQ, 256, 0, stream>>>(attn, cosb, ref);

  // h = attn * V per (b,h): [256x256] x [256x32]
  launch_gemm(stream, attn, LQ, SBH, LLQQ, Vh, BZ * DM, DM, HD, nullptr,
              rel, BZ * DM, DM, HD, LQ, HD, LQ, BZ * NH, NH, 1, 0);

  // tgt2 = h + q
  add_inplace_k<<<8192, 256, 0, stream>>>(rel, q, (long)NQROWS * DM);

  // x1 = LN(tgt + tgt2)
  ln_add_k<<<NQROWS, 256, 0, stream>>>(tgt, rel, g1, be1, x1);

  // ---- deformable attention ---------------------------------------------
  // value projection (dominant GEMM, 131072 rows)
  launch_gemm(stream, memory, DM, 0, 0, Wvp, DM, 0, 0, bvp, vproj, DM, 0, 0,
              NVROWS, DM, DM, 1, 1, 0, 0);
  vzero_pad_k<<<32768, 256, 0, stream>>>(vproj, padmask, (long)NVROWS * DM);

  // sampling offsets / weights from tgt2
  launch_gemm(stream, rel, DM, 0, 0, Woff, DM, 0, 0, boff, offr, NH * NK, 0, 0,
              NQROWS, NH * NK, DM, 1, 1, 0, 0);
  launch_gemm(stream, rel, DM, 0, 0, Waw, DM, 0, 0, baw, awr, NH * NK, 0, 0,
              NQROWS, NH * NK, DM, 1, 1, 0, 0);

  deform_sample_k<<<BZ * NH * LQ, 32, 0, stream>>>(vproj, offr, awr, ref, samp);

  // output projection
  launch_gemm(stream, samp, DM, 0, 0, Wop, DM, 0, 0, bop, t2c, DM, 0, 0,
              NQROWS, DM, DM, 1, 1, 0, 0);

  // x2 = LN(x1 + tgt2c)
  ln_add_k<<<NQROWS, 256, 0, stream>>>(x1, t2c, g2, be2, x2);

  // ---- FFN ---------------------------------------------------------------
  launch_gemm(stream, x2, DM, 0, 0, W1, DM, 0, 0, bf1, hid, DFF, 0, 0,
              NQROWS, DFF, DM, 1, 1, 0, 1);          // ReLU fused
  launch_gemm(stream, hid, DFF, 0, 0, W2, DFF, 0, 0, bf2, ffn, DM, 0, 0,
              NQROWS, DM, DFF, 1, 1, 0, 0);

  // out = LN(x2 + ffn)
  ln_add_k<<<NQROWS, 256, 0, stream>>>(x2, ffn, g3, be3, out);
}
